// AnchorStore_53102975647798
// MI455X (gfx1250) — matrix operand order repaired and checked by reading, not compiled, not run on hardware
//
#include <hip/hip_runtime.h>
#include <hip/hip_bf16.h>

// ---------------------------------------------------------------------------
// KL-nearest-anchor classifier for MI455X (gfx1250, wave32, WMMA).
//   score[k][b] = sum_d a[k,d]*log2(a[k,d])  -  sum_d a[k,d]*log2(q[b,d])
// argmin_k score is identical to the reference argmin (positive scaling:
// ln2 and 1/D factors drop out), so v_log_f32 (log2) is used everywhere.
// Cross term = 2048x256x50257 GEMM on v_wmma_f32_16x16x32_bf16 with an
// a = hi + lo bf16 split (two WMMAs, near-fp32 accuracy, still HBM-bound).
// Double-buffered LDS staging: one barrier per 32-d chunk, next chunk's
// global loads overlap the WMMA block. A-pad value is 1.0f so both the
// self term (1*log2(1)=0) and the cross term (1 * b_pad=0) stay exact
// without any per-element masking in the hot loop.
// ---------------------------------------------------------------------------

typedef __attribute__((ext_vector_type(16))) __bf16 v16bf;
typedef __attribute__((ext_vector_type(8)))  float  v8f;

#define K_ANCH   2048
#define D_DIM    50257
#define B_Q      256
#define CHUNK    32
#define NCHUNK   1571                 // ceil(D/32)
#define DP       (NCHUNK * CHUNK)     // 50272, zero-padded depth
#define SPLITS   16                   // split-K over D
#define MTILE    64                   // k-rows per workgroup
#define LDSP     48                   // LDS row pitch in bf16 elems (96 B)

__device__ __forceinline__ unsigned pk2(__bf16 a, __bf16 b) {
    union { __bf16 h[2]; unsigned u; } x;
    x.h[0] = a; x.h[1] = b;
    return x.u;
}

// ---------------- kernel 1: logqT[b][dp] = bf16(log2(query[b][d])) ----------
__global__ void prep_logq(const float* __restrict__ q, __bf16* __restrict__ lq) {
    size_t flat = ((size_t)blockIdx.x * blockDim.x + threadIdx.x) * 4;
    if (flat >= (size_t)B_Q * DP) return;
    int b   = (int)(flat / DP);
    int col = (int)(flat % DP);            // DP % 4 == 0 -> same row
    const float* qrow = q + (size_t)b * D_DIM;
    __bf16 o[4];
#pragma unroll
    for (int j = 0; j < 4; ++j) {
        int d = col + j;
        float v = (d < D_DIM) ? __log2f(qrow[d]) : 0.0f;
        o[j] = (__bf16)v;
    }
    *(uint2*)(lq + (size_t)b * DP + col) = make_uint2(pk2(o[0], o[1]), pk2(o[2], o[3]));
}

// ---------------- kernel 2: fused self-term + WMMA cross GEMM ---------------
__global__ __launch_bounds__(512)
void kl_gemm(const float* __restrict__ A, const __bf16* __restrict__ lq,
             float* __restrict__ cross, float* __restrict__ selfsum) {
    // ping-pong buffers: 2 x (hi + lo) x 64 rows x 48 pitch x 2B = 24 KB
    __shared__ alignas(16) __bf16 sh_hi[2][MTILE * LDSP];
    __shared__ alignas(16) __bf16 sh_lo[2][MTILE * LDSP];

    const int tid  = threadIdx.x;
    const int wave = tid >> 5;             // 0..15 -> owns b-cols 16*wave..+15
    const int lane = tid & 31;
    const int ng   = lane >> 4;            // lane group (K-dim half)
    const int nl   = lane & 15;            // N within tile / M within subtile

    const int k0      = blockIdx.x * MTILE;
    const int c_begin = (int)(( (long)blockIdx.y      * NCHUNK) / SPLITS);
    const int c_end   = (int)((((long)blockIdx.y + 1) * NCHUNK) / SPLITS);

    // staging role: 8 threads per A row, 4 elems each (row stride is odd ->
    // scalar b32 loads; cacheline utilization is unaffected)
    const int srow = tid >> 3;             // 0..63
    const int scol = (tid & 7) * 4;        // 0,4,...,28
    const float* arow = A + (size_t)(k0 + srow) * D_DIM;

    // B fragment source: lane nl -> column b = 16*wave + nl, contiguous d's
    const __bf16* brow = lq + (size_t)(wave * 16 + nl) * DP;

    float selfacc = 0.0f;
    v8f acc[4];
#pragma unroll
    for (int mt = 0; mt < 4; ++mt)
#pragma unroll
        for (int r = 0; r < 8; ++r) acc[mt][r] = 0.0f;

    // chunk loader: uniform SCC branch, guards only the single partial chunk;
    // pad with 1.0f (self: 1*log2(1)=0; cross: 1 * b_pad(0) = 0)
    auto load4 = [&](int c, float va[4]) {
        const int base = c * CHUNK + scol;
        if (c != NCHUNK - 1) {
#pragma unroll
            for (int j = 0; j < 4; ++j) va[j] = arow[base + j];
        } else {
#pragma unroll
            for (int j = 0; j < 4; ++j)
                va[j] = (base + j < D_DIM) ? arow[base + j] : 1.0f;
        }
    };

    float va[4];
    load4(c_begin, va);

    for (int c = c_begin; c < c_end; ++c) {
        const int buf = (c - c_begin) & 1;

        // ---- self-term + (hi, lo) bf16 split, store into current buffer ----
#pragma unroll
        for (int j = 0; j < 4; ++j)
            selfacc = __builtin_fmaf(va[j], __log2f(va[j]), selfacc);
        __bf16 hi[4], lo[4];
#pragma unroll
        for (int j = 0; j < 4; ++j) {
            hi[j] = (__bf16)va[j];
            lo[j] = (__bf16)(va[j] - (float)hi[j]);
        }
        *(uint2*)(&sh_hi[buf][srow * LDSP + scol]) =
            make_uint2(pk2(hi[0], hi[1]), pk2(hi[2], hi[3]));
        *(uint2*)(&sh_lo[buf][srow * LDSP + scol]) =
            make_uint2(pk2(lo[0], lo[1]), pk2(lo[2], lo[3]));
        __syncthreads();                    // one barrier per chunk

        // ---- issue next chunk's global loads; they fly during the WMMAs ----
        float vn[4] = { va[0], va[1], va[2], va[3] };
        if (c + 1 < c_end) load4(c + 1, vn);
        if (c + 2 < c_end)                  // global_prefetch_b8 for c+2
            __builtin_prefetch(arow + (size_t)(c + 2) * CHUNK + scol, 0, 1);

        // ---- B fragment: 32x16 bf16; lanes 0-15 hold K=0..15, 16-31 K=16..31
        union { uint4 u[2]; v16bf v; } bf;
        const __bf16* bp = brow + (size_t)c * CHUNK + ng * 16;
        bf.u[0] = *(const uint4*)(bp);
        bf.u[1] = *(const uint4*)(bp + 8);

        // ---- 4 M-subtiles x (hi + lo) WMMAs from the current buffer ----
#pragma unroll
        for (int mt = 0; mt < 4; ++mt) {
            const int m = mt * 16 + nl;
            union { uint4 u[2]; v16bf v; } ah, al;
            // A 16x32 bf16 layout: VGPR0-7 <- K=g*16..+15 (matches B)
            ah.u[0] = *(const uint4*)(&sh_hi[buf][m * LDSP + ng * 16]);
            ah.u[1] = *(const uint4*)(&sh_hi[buf][m * LDSP + ng * 16 + 8]);
            al.u[0] = *(const uint4*)(&sh_lo[buf][m * LDSP + ng * 16]);
            al.u[1] = *(const uint4*)(&sh_lo[buf][m * LDSP + ng * 16 + 8]);
            acc[mt] = __builtin_amdgcn_wmma_f32_16x16x32_bf16(
                false, ah.v, false, bf.v, (short)0, acc[mt], false, false);
            acc[mt] = __builtin_amdgcn_wmma_f32_16x16x32_bf16(
                false, al.v, false, bf.v, (short)0, acc[mt], false, false);
        }

#pragma unroll
        for (int j = 0; j < 4; ++j) va[j] = vn[j];
    }

    // ---- epilogue: accumulate split-K partials ----
    const int bb = wave * 16 + nl;
#pragma unroll
    for (int mt = 0; mt < 4; ++mt)
#pragma unroll
        for (int r = 0; r < 8; ++r) {
            const int kk = k0 + mt * 16 + r + 8 * ng;   // C layout: M = r + 8*(lane/16)
            unsafeAtomicAdd(&cross[(size_t)kk * B_Q + bb], acc[mt][r]);
        }
    unsafeAtomicAdd(&selfsum[k0 + srow], selfacc);
}

// ---------------- kernel 3: per-query argmin + label lookup -----------------
__global__ void argmin_label(const float* __restrict__ cross,
                             const float* __restrict__ selfsum,
                             const int* __restrict__ labels,
                             float* __restrict__ out) {
    const int b = blockIdx.x;
    const int t = threadIdx.x;
    float best = 3.4e38f;
    int   bi   = K_ANCH;
    for (int k = t; k < K_ANCH; k += 256) {
        float s = selfsum[k] - cross[(size_t)k * B_Q + b];
        if (s < best) { best = s; bi = k; }
    }
    __shared__ float vals[256];
    __shared__ int   idxs[256];
    vals[t] = best; idxs[t] = bi;
    __syncthreads();
    for (int off = 128; off > 0; off >>= 1) {
        if (t < off) {
            if (vals[t + off] < vals[t] ||
                (vals[t + off] == vals[t] && idxs[t + off] < idxs[t])) {
                vals[t] = vals[t + off];
                idxs[t] = idxs[t + off];
            }
        }
        __syncthreads();
    }
    if (t == 0) out[b] = (float)labels[idxs[0]];
}

// ---------------------------------------------------------------------------
extern "C" void kernel_launch(void* const* d_in, const int* in_sizes, int n_in,
                              void* d_out, int out_size, void* d_ws, size_t ws_size,
                              hipStream_t stream) {
    const float* query  = (const float*)d_in[0];   // [B, D] fp32
    const float* anchor = (const float*)d_in[1];   // [K, D] fp32
    const int*   labels = (const int*)d_in[2];     // [K]
    float*       out    = (float*)d_out;           // [B]

    char* ws = (char*)d_ws;
    float*  cross   = (float*)ws;                                   // K*B fp32
    float*  selfsum = (float*)(ws + (size_t)K_ANCH * B_Q * 4);      // K fp32
    __bf16* lq      = (__bf16*)(ws + (size_t)K_ANCH * B_Q * 4 + (size_t)K_ANCH * 4);

    // zero the split-K accumulators (graph-capture safe)
    hipMemsetAsync(ws, 0, (size_t)K_ANCH * B_Q * 4 + (size_t)K_ANCH * 4, stream);

    // 1) log2(query) -> padded bf16 B-matrix
    {
        size_t quads = (size_t)B_Q * DP / 4;
        int blocks = (int)((quads + 255) / 256);
        prep_logq<<<blocks, 256, 0, stream>>>(query, lq);
    }
    // 2) fused self-term + WMMA cross GEMM (32 k-tiles x 16 D-splits)
    kl_gemm<<<dim3(K_ANCH / MTILE, SPLITS), 512, 0, stream>>>(anchor, lq, cross, selfsum);
    // 3) argmin over K per query, emit label as float
    argmin_label<<<B_Q, 256, 0, stream>>>(cross, selfsum, labels, out);
}
